// IntegratedNeuralBrain_34677565948788
// MI455X (gfx1250) — matrix operand order repaired and checked by reading, not compiled, not run on hardware
//
#include <hip/hip_runtime.h>

typedef __attribute__((ext_vector_type(16))) __bf16 v16bf;
typedef __attribute__((ext_vector_type(8)))  __bf16 v8bf;
typedef __attribute__((ext_vector_type(4)))  __bf16 v4bf;
typedef __attribute__((ext_vector_type(8)))  float  v8f;
typedef __attribute__((ext_vector_type(4)))  float  f4v;

#define SEQ    2048
#define DMODEL 768
#define HEADS  8
#define HD     96
#define MROWS  4096      // B*S
#define THREED 2304
#define DBRAIN 1024
#define CACHE_ELEMS 201326592LL   // 4*4096*16*8*96

union FragU { v16bf v; v8bf h[2]; };
union PackU { v16bf v; __bf16 e[16]; };

__device__ __forceinline__ v16bf make_frag(v8bf lo, v8bf hi) {
  FragU u; u.h[0] = lo; u.h[1] = hi; return u.v;
}

// ---------------------------------------------------------------- cvt f32->bf16 (x4)
__global__ void nb_cvt_bf16(const float* __restrict__ src, __bf16* __restrict__ dst, int n4) {
  int i = blockIdx.x * blockDim.x + threadIdx.x;
  if (i < n4) {
    f4v v = *(const f4v*)(src + (size_t)i * 4);
    *(v4bf*)(dst + (size_t)i * 4) = __builtin_convertvector(v, v4bf);
  }
}

// ---------------------------------------------------------------- generic bf16 WMMA GEMM
// C[M,N] = A[M,K] @ B[K,N]; A,B bf16 row-major.
// MODE 0: f32 out.  MODE 1: bf16 out.  MODE 2: f32 out + bias.
// Block: 256 thr (8 waves), tile 128x128x32, double-buffered LDS. Waves 2(M)x4(N), 64x32 each.
template <int MODE>
__global__ __launch_bounds__(256)
void nb_gemm_bf16(const __bf16* __restrict__ A, const __bf16* __restrict__ Bm,
                  void* __restrict__ Cout, const float* __restrict__ bias,
                  int M, int N, int K) {
  __shared__ __align__(16) __bf16 As[2][128 * 40];   // row stride 40 bf16 (80B)
  __shared__ __align__(16) __bf16 Bs[2][128 * 40];   // transposed: [n][k]

  const int t    = threadIdx.x;
  const int lane = t & 31;
  const int w    = t >> 5;
  const int wm   = w & 1;        // wave row -> 64 rows
  const int wn   = w >> 1;       // wave col -> 32 cols
  const int hb   = lane >> 4;    // wave half
  const int ln   = lane & 15;
  const int blockRow = blockIdx.y * 128;
  const int blockCol = blockIdx.x * 128;

  v8f acc[4][2];
#pragma unroll
  for (int m = 0; m < 4; ++m)
#pragma unroll
    for (int n = 0; n < 2; ++n)
#pragma unroll
      for (int i = 0; i < 8; ++i) acc[m][n][i] = 0.0f;

  // cooperative-load assignments
  const int arow = t >> 1, ahalf = t & 1;          // A: 128 rows x 2 half-rows (16 bf16)
  const int bkp  = t >> 4, bnb   = (t & 15) << 3;  // B: 16 k-pairs x 16 col-groups (8 cols)

  // ---- prologue: stage tile 0
  {
    const __bf16* ag = A + (size_t)(blockRow + arow) * K + ahalf * 16;
    v8bf a0 = *(const v8bf*)ag;
    v8bf a1 = *(const v8bf*)(ag + 8);
    const __bf16* bg = Bm + (size_t)(2 * bkp) * N + blockCol + bnb;
    v8bf b0 = *(const v8bf*)bg;
    v8bf b1 = *(const v8bf*)(bg + N);
    *(v8bf*)&As[0][arow * 40 + ahalf * 16]     = a0;
    *(v8bf*)&As[0][arow * 40 + ahalf * 16 + 8] = a1;
#pragma unroll
    for (int j = 0; j < 8; ++j) {
      union { __bf16 h[2]; unsigned u; } pk;
      pk.h[0] = b0[j]; pk.h[1] = b1[j];
      *(unsigned*)&Bs[0][(bnb + j) * 40 + 2 * bkp] = pk.u;   // adjacent k's contiguous
    }
  }
  __syncthreads();

  const int nk = K >> 5;
  for (int kb = 0; kb < nk; ++kb) {
    const int cur = kb & 1;
    const bool more = (kb + 1) < nk;

    // prefetch next tile into registers (overlaps with WMMA below)
    v8bf na0 = {}, na1 = {}, nb0 = {}, nb1 = {};
    if (more) {
      const __bf16* ag = A + (size_t)(blockRow + arow) * K + ((kb + 1) << 5) + ahalf * 16;
      na0 = *(const v8bf*)ag;
      na1 = *(const v8bf*)(ag + 8);
      const __bf16* bg = Bm + (size_t)(((kb + 1) << 5) + 2 * bkp) * N + blockCol + bnb;
      nb0 = *(const v8bf*)bg;
      nb1 = *(const v8bf*)(bg + N);
    }

    // compute from current buffer
    const __bf16* asrc = As[cur];
    const __bf16* bsrc = Bs[cur];
    v16bf af[4];
#pragma unroll
    for (int m = 0; m < 4; ++m) {
      int r = wm * 64 + m * 16 + ln;
      af[m] = make_frag(*(const v8bf*)&asrc[r * 40 + hb * 8],
                        *(const v8bf*)&asrc[r * 40 + 16 + hb * 8]);
    }
    v16bf bf[2];
#pragma unroll
    for (int n = 0; n < 2; ++n) {
      int c = wn * 32 + n * 16 + ln;
      bf[n] = make_frag(*(const v8bf*)&bsrc[c * 40 + hb * 8],
                        *(const v8bf*)&bsrc[c * 40 + 16 + hb * 8]);
    }
#pragma unroll
    for (int m = 0; m < 4; ++m)
#pragma unroll
      for (int n = 0; n < 2; ++n)
        acc[m][n] = __builtin_amdgcn_wmma_f32_16x16x32_bf16(
            false, af[m], false, bf[n], (short)0, acc[m][n], false, false);

    // stage next tile into the other buffer
    if (more) {
      const int nxt = cur ^ 1;
      *(v8bf*)&As[nxt][arow * 40 + ahalf * 16]     = na0;
      *(v8bf*)&As[nxt][arow * 40 + ahalf * 16 + 8] = na1;
#pragma unroll
      for (int j = 0; j < 8; ++j) {
        union { __bf16 h[2]; unsigned u; } pk;
        pk.h[0] = nb0[j]; pk.h[1] = nb1[j];
        *(unsigned*)&Bs[nxt][(bnb + j) * 40 + 2 * bkp] = pk.u;
      }
    }
    __syncthreads();
  }

  // ---- straight-line epilogue (compile-time mode)
  float* Cf  = (float*)Cout;
  __bf16* Cb = (__bf16*)Cout;
#pragma unroll
  for (int m = 0; m < 4; ++m)
#pragma unroll
    for (int n = 0; n < 2; ++n) {
      int col = blockCol + wn * 32 + n * 16 + ln;
      float bv = 0.0f;
      if constexpr (MODE == 2) bv = bias[col];
#pragma unroll
      for (int i = 0; i < 8; ++i) {
        int row = blockRow + wm * 64 + m * 16 + i + 8 * hb;
        float vv = acc[m][n][i] + bv;
        if constexpr (MODE == 1) Cb[(size_t)row * N + col] = (__bf16)vv;
        else                     Cf[(size_t)row * N + col] = vv;
      }
    }
}

// ---------------------------------------------------------------- qkv split: q(scaled)/k [BH,S,hd], vT [BH,hd,S]
__global__ void nb_split_qkv(const float* __restrict__ qkv, __bf16* __restrict__ qb,
                             __bf16* __restrict__ kb, __bf16* __restrict__ vt) {
  int i = blockIdx.x * blockDim.x + threadIdx.x;     // over 2*8*2048*96
  if (i >= 2 * HEADS * SEQ * HD) return;
  int d  = i % HD;
  int s  = (i / HD) % SEQ;
  int bh = i / (HD * SEQ);
  int b = bh >> 3, hh = bh & 7;
  const float* base = qkv + (size_t)(b * SEQ + s) * THREED + hh * HD + d;
  qb[i] = (__bf16)(base[0] * 0.10206207261596575f);  // 1/sqrt(96)
  kb[i] = (__bf16)base[DMODEL];
  vt[(size_t)bh * HD * SEQ + (size_t)d * SEQ + s] = (__bf16)base[2 * DMODEL];
}

// ---------------------------------------------------------------- flash attention, 1 wave per 16 queries
// Scores bounce through LDS once (f32, [row][key]); softmax runs in A-frag order so the
// probability fragment is packed directly in registers (no second LDS trip, no butterfly
// bpermute reductions -- one xor-16 shuffle per reduction).
__global__ __launch_bounds__(32)
void nb_flash_attn(const __bf16* __restrict__ q, const __bf16* __restrict__ k,
                   const __bf16* __restrict__ vt, __bf16* __restrict__ ctx) {
  __shared__ __align__(16) float Sld[16 * 32];   // 2 KB raw-score tile
  const int lane = threadIdx.x;
  const int hb = lane >> 4, ln = lane & 15;
  const int bh = blockIdx.x >> 7;
  const int q0 = (blockIdx.x & 127) << 4;
  const int b = bh >> 3, hh = bh & 7;
  const __bf16* qp = q  + (size_t)bh * SEQ * HD;
  const __bf16* kp = k  + (size_t)bh * SEQ * HD;
  const __bf16* vp = vt + (size_t)bh * HD * SEQ;

  v16bf qf[3];
#pragma unroll
  for (int c = 0; c < 3; ++c) {
    const __bf16* p = qp + (size_t)(q0 + ln) * HD + c * 32;
    qf[c] = make_frag(*(const v8bf*)(p + hb * 16), *(const v8bf*)(p + hb * 16 + 8));
  }

  v8f accv[6];
#pragma unroll
  for (int tt = 0; tt < 6; ++tt)
#pragma unroll
    for (int i = 0; i < 8; ++i) accv[tt][i] = 0.0f;
  float mi_l = -1e30f, li_l = 0.0f;   // stats for row `ln` (duplicated in both halves)

  for (int kc = 0; kc < SEQ / 32; ++kc) {
    const int kbase = kc * 32;
    v8f s0, s1;
#pragma unroll
    for (int i = 0; i < 8; ++i) { s0[i] = 0.0f; s1[i] = 0.0f; }
#pragma unroll
    for (int c = 0; c < 3; ++c) {
      const __bf16* p0 = kp + (size_t)(kbase + ln) * HD + c * 32 + hb * 16;
      v16bf bf0 = make_frag(*(const v8bf*)p0, *(const v8bf*)(p0 + 8));
      s0 = __builtin_amdgcn_wmma_f32_16x16x32_bf16(false, qf[c], false, bf0, (short)0, s0, false, false);
      const __bf16* p1 = kp + (size_t)(kbase + 16 + ln) * HD + c * 32 + hb * 16;
      v16bf bf1 = make_frag(*(const v8bf*)p1, *(const v8bf*)(p1 + 8));
      s1 = __builtin_amdgcn_wmma_f32_16x16x32_bf16(false, qf[c], false, bf1, (short)0, s1, false, false);
    }

    // scatter raw scores: C-frag -> LDS [row][key] f32
#pragma unroll
    for (int i = 0; i < 8; ++i) {
      Sld[(i + 8 * hb) * 32 + ln]      = s0[i];
      Sld[(i + 8 * hb) * 32 + 16 + ln] = s1[i];
    }
    __syncthreads();
    // gather row `ln` in A-frag element order: keys {hb*8..+7} U {16+hb*8..+7}
    float p[16];
    {
      f4v r0 = *(const f4v*)&Sld[ln * 32 + hb * 8];
      f4v r1 = *(const f4v*)&Sld[ln * 32 + hb * 8 + 4];
      f4v r2 = *(const f4v*)&Sld[ln * 32 + 16 + hb * 8];
      f4v r3 = *(const f4v*)&Sld[ln * 32 + 16 + hb * 8 + 4];
#pragma unroll
      for (int j = 0; j < 4; ++j) {
        p[j] = r0[j]; p[4 + j] = r1[j]; p[8 + j] = r2[j]; p[12 + j] = r3[j];
      }
    }
    __syncthreads();

    // row max: 15 local max + one xor-16 shuffle
    float lm = p[0];
#pragma unroll
    for (int j = 1; j < 16; ++j) lm = fmaxf(lm, p[j]);
    lm = fmaxf(lm, __shfl_xor(lm, 16, 32));
    float mnew = fmaxf(mi_l, lm);
    float alpha = __expf(mi_l - mnew);

    // exponentiate in place, local sum + one xor-16 shuffle
    float ls = 0.0f;
#pragma unroll
    for (int j = 0; j < 16; ++j) { p[j] = __expf(p[j] - mnew); ls += p[j]; }
    ls += __shfl_xor(ls, 16, 32);
    li_l = li_l * alpha + ls;
    mi_l = mnew;

    // pack probability A-fragment directly in registers
    PackU pu;
#pragma unroll
    for (int j = 0; j < 16; ++j) pu.e[j] = (__bf16)p[j];

    // broadcast alpha into C-frag row order (8 shuffles) and rescale accumulators
    float a8[8];
#pragma unroll
    for (int i = 0; i < 8; ++i) a8[i] = __shfl(alpha, i + 8 * hb, 32);
#pragma unroll
    for (int tt = 0; tt < 6; ++tt)
#pragma unroll
      for (int i = 0; i < 8; ++i) accv[tt][i] *= a8[i];

    // PV accumulate
#pragma unroll
    for (int tt = 0; tt < 6; ++tt) {
      const __bf16* vv = vp + (size_t)(tt * 16 + ln) * SEQ + kbase;
      v16bf vf = make_frag(*(const v8bf*)(vv + hb * 8), *(const v8bf*)(vv + 16 + hb * 8));
      accv[tt] = __builtin_amdgcn_wmma_f32_16x16x32_bf16(false, pu.v, false, vf, (short)0, accv[tt], false, false);
    }
  }

  // final normalization: broadcast 1/l into C-frag row order
  float invl = 1.0f / li_l;
  float inv8[8];
#pragma unroll
  for (int i = 0; i < 8; ++i) inv8[i] = __shfl(invl, i + 8 * hb, 32);
#pragma unroll
  for (int tt = 0; tt < 6; ++tt)
#pragma unroll
    for (int i = 0; i < 8; ++i) {
      int srow = q0 + i + 8 * hb;
      ctx[((size_t)(b * SEQ + srow)) * DMODEL + hh * HD + tt * 16 + ln] =
          (__bf16)(accv[tt][i] * inv8[i]);
    }
}

// ---------------------------------------------------------------- fused KV-cache fill (zero + scatter)
// grid.x = 4*4096*16 rows of 768 floats; grid.y = {k,v}; 192 thr x float4; NT stores.
__global__ __launch_bounds__(192)
void nb_kv_cache(const float* __restrict__ qkv, float* __restrict__ cache) {
  const int row   = blockIdx.x;          // (layer, blk, pos)
  const int kv    = blockIdx.y;
  const int pos   = row & 15;
  const int blk   = (row >> 4) & 4095;
  const int layer = row >> 16;
  const int col4  = threadIdx.x << 2;
  f4v val = {0.f, 0.f, 0.f, 0.f};
  const int rel = blk - layer * 128;     // n_blocks = 128 per layer
  if ((unsigned)rel < 128u) {
    const int s = rel * 16 + pos;        // batch 0 only
    val = *(const f4v*)(qkv + (size_t)s * THREED + DMODEL * (1 + kv) + col4);
  }
  __builtin_nontemporal_store(
      val, (f4v*)(cache + (size_t)kv * CACHE_ELEMS + (size_t)row * 768 + col4));
}

// ---------------------------------------------------------------- launch
extern "C" void kernel_launch(void* const* d_in, const int* in_sizes, int n_in,
                              void* d_out, int out_size, void* d_ws, size_t ws_size,
                              hipStream_t stream) {
  (void)in_sizes; (void)n_in; (void)out_size; (void)ws_size;
  const float* hs    = (const float*)d_in[0];   // [2,2048,768]
  const float* wqkv  = (const float*)d_in[1];   // [768,2304]
  const float* wao   = (const float*)d_in[2];   // [768,768]
  const float* wproj = (const float*)d_in[3];   // [768,1024]
  const float* bproj = (const float*)d_in[4];   // [1024]
  float* out = (float*)d_out;                   // [out | k_cache | v_cache]

  char* ws = (char*)d_ws;
  float*  qkv_f  = (float*) (ws + 0);           // 37,748,736 B  (4096x2304 f32)
  __bf16* hs_b   = (__bf16*)(ws + 37748736);    //  6,291,456 B
  __bf16* wqkv_b = (__bf16*)(ws + 44040192);    //  3,538,944 B
  __bf16* wao_b  = (__bf16*)(ws + 47579136);    //  1,179,648 B
  __bf16* wpr_b  = (__bf16*)(ws + 48758784);    //  1,572,864 B
  __bf16* q_b    = (__bf16*)(ws + 50331648);    //  6,291,456 B
  __bf16* k_b    = (__bf16*)(ws + 56623104);    //  6,291,456 B
  __bf16* vt_b   = (__bf16*)(ws + 62914560);    //  6,291,456 B
  __bf16* ctx_b  = (__bf16*)(ws + 69206016);    //  6,291,456 B
  __bf16* ao_b   = (__bf16*)(ws + 75497472);    //  6,291,456 B  -> total 81.8 MB

  // precision downcasts (x4 vectorized)
  nb_cvt_bf16<<<3145728 / 4 / 256, 256, 0, stream>>>(hs,    hs_b,   3145728 / 4);
  nb_cvt_bf16<<<1769472 / 4 / 256, 256, 0, stream>>>(wqkv,  wqkv_b, 1769472 / 4);
  nb_cvt_bf16<<< 589824 / 4 / 256, 256, 0, stream>>>(wao,   wao_b,   589824 / 4);
  nb_cvt_bf16<<< 786432 / 4 / 256, 256, 0, stream>>>(wproj, wpr_b,   786432 / 4);

  // qkv = hs @ w_qkv   (f32 out, needed for the cache)
  nb_gemm_bf16<0><<<dim3(THREED / 128, MROWS / 128), 256, 0, stream>>>(
      hs_b, wqkv_b, (void*)qkv_f, nullptr, MROWS, THREED, DMODEL);

  // KV cache: single pass over the dominant 1.61 GB output
  nb_kv_cache<<<dim3(4 * 4096 * 16, 2), 192, 0, stream>>>(qkv_f, out + 4194304);

  // split q/k/vT for attention
  nb_split_qkv<<<3145728 / 256, 256, 0, stream>>>(qkv_f, q_b, k_b, vt_b);

  // flash attention: 16 bh-pairs x 128 query tiles
  nb_flash_attn<<<2 * HEADS * (SEQ / 16), 32, 0, stream>>>(q_b, k_b, vt_b, ctx_b);

  // attn_out = ctx @ w_attn_out  (bf16 out)
  nb_gemm_bf16<1><<<dim3(DMODEL / 128, MROWS / 128), 256, 0, stream>>>(
      ctx_b, wao_b, (void*)ao_b, nullptr, MROWS, DMODEL, DMODEL);

  // out = attn_out @ w_proj + b_proj  (f32 out)
  nb_gemm_bf16<2><<<dim3(DBRAIN / 128, MROWS / 128), 256, 0, stream>>>(
      ao_b, wpr_b, (void*)out, bproj, MROWS, DBRAIN, DMODEL);
}
